// FAST_3770981286528
// MI455X (gfx1250) — matrix operand order repaired and checked by reading, not compiled, not run on hardware
//
#include <hip/hip_runtime.h>
#include <hip/hip_bf16.h>

// Problem constants
#define S_  512
#define L_  5
#define T_  20
#define E_  768
#define H_  64
#define G_  256   // 4H
#define KC  128   // K-chunk for GEMM staging
#define MPAD 112  // 100 rows padded to 7*16
#define NTHREADS 256

typedef __bf16 bf16_t;
typedef __attribute__((ext_vector_type(16))) __bf16 v16bf;
typedef __attribute__((ext_vector_type(8)))  float  v8f;

// ---------------- LDS layout (bytes) ----------------
#define OFF_U      0
#define SZ_U       (MPAD*G_*4)            // 114688: u[112][256] f32
#define OFF_BIG    (OFF_U + SZ_U)
// Phase 1: fragment-major staging buffers
//   A frags: [kk(4)][mt(7)][lane(32)][16 bf16]  = 28672 B
//   B frags: [kk(4)][nt(16)][lane(32)][16 bf16] = 65536 B
#define OFF_AS     OFF_BIG
#define SZ_AS      (4*7*32*16*2)           // 28672
#define OFF_BS     (OFF_AS + SZ_AS)
#define SZ_BS      (4*16*32*16*2)          // 65536
#define SZ_BIG     (SZ_AS + SZ_BS)         // 94208 (reused by Wall/Wd, then Wih)
#define OFF_WALL   OFF_BIG                 // f32 [64][256]
#define OFF_WD     (OFF_BIG + H_*G_*4)     // f32 [64][64]
#define OFF_WIH    OFF_BIG                 // f32 [64][256]
#define OFF_W1     (OFF_BIG + SZ_BIG)      // f32 [64][64]
#define OFF_W2     (OFF_W1 + H_*H_*4)      // f32 [64][64]
#define OFF_FULL   (OFF_W2 + H_*H_*4)      // f32 [20][5][64] (o-gate sequence)
#define OFF_H      (OFF_FULL + T_*L_*H_*4)
#define OFF_C      (OFF_H  + L_*H_*4)
#define OFF_CS1    (OFF_C  + L_*H_*4)
#define OFF_GATES  (OFF_CS1+ L_*H_*4)      // f32 [5][256]
#define OFF_TS     (OFF_GATES + L_*G_*4)   // f32 [5][20] (padded to 512B)
#define OFF_BWALL  (OFF_TS + 512)
#define OFF_BWD    (OFF_BWALL + G_*4)
#define OFF_BU     (OFF_BWD + H_*4)
#define OFF_B1     (OFF_BU + G_*4)
#define OFF_B2     (OFF_B1 + H_*4)
#define OFF_V      (OFF_B2 + H_*4)
#define OFF_W1H    (OFF_V + H_*4)          // f32 [5][64]
#define OFF_SCORE  (OFF_W1H + L_*H_*4)     // f32 [5][20] (padded 512B), reused as aw
#define OFF_TVEC   (OFF_SCORE + 512)       // f32 [5][64]
#define OFF_HD     (OFF_TVEC + L_*H_*4)    // f32 [5][64]
#define OFF_S2     (OFF_HD + L_*H_*4)      // 16 floats: s2[5], aw2 at +8
#define OFF_BSUM   (OFF_S2 + 64)           // f32 [256]
#define OFF_FT     (OFF_BSUM + G_*4)       // f32 [64]
#define SMEM_BYTES (OFF_FT + H_*4)         // ~285.5 KB < 320 KB

__device__ __forceinline__ float sigm(float x) { return 1.f / (1.f + __expf(-x)); }

// round-to-nearest-even f32 -> bf16 bits
__device__ __forceinline__ unsigned bf16_bits(float f) {
  union { float f; unsigned u; } x; x.f = f;
  unsigned rb = ((x.u >> 16) & 1u) + 0x7FFFu;
  return (x.u + rb) >> 16;
}
__device__ __forceinline__ unsigned pack_bf16x2(float lo, float hi) {
  return bf16_bits(lo) | (bf16_bits(hi) << 16);
}

__global__ __launch_bounds__(NTHREADS)
void athena_stocknet_kernel(
    const float* __restrict__ p_x,     // [S,L,T,E]
    const float* __restrict__ p_ts,    // [S,L,T]
    const float* __restrict__ p_Wall,  // [S,H,4H]
    const float* __restrict__ p_bWall, // [S,4H]
    const float* __restrict__ p_Uall,  // [S,E,4H]
    const float* __restrict__ p_bUall, // [S,4H]
    const float* __restrict__ p_Wd,    // [S,H,H]
    const float* __restrict__ p_bWd,   // [S,H]
    const float* __restrict__ p_W1,  const float* __restrict__ p_b1,
    const float* __restrict__ p_W2,  const float* __restrict__ p_b2,
    const float* __restrict__ p_V,   const float* __restrict__ p_bV,
    const float* __restrict__ p_Wih, const float* __restrict__ p_bih,
    const float* __restrict__ p_bhh,
    const float* __restrict__ p_linw, const float* __restrict__ p_linb,
    float* __restrict__ p_out)
{
  extern __shared__ char smem[];
  const int s    = blockIdx.x;
  const int tid  = threadIdx.x;
  const int lane = tid & 31;
  const int wave = tid >> 5;

  float*    u_s   = (float*)(smem + OFF_U);
  unsigned* As32  = (unsigned*)(smem + OFF_AS);
  unsigned* Bs32  = (unsigned*)(smem + OFF_BS);
  const v16bf* Afrag = (const v16bf*)(smem + OFF_AS);
  const v16bf* Bfrag = (const v16bf*)(smem + OFF_BS);
  float*    bU_s  = (float*)(smem + OFF_BU);

  const float* xS = p_x    + (size_t)s * (L_*T_*E_);
  const float* uS = p_Uall + (size_t)s * (E_*G_);

  for (int i = tid; i < G_; i += NTHREADS) bU_s[i] = p_bUall[s*G_ + i];

  // ---------------- Phase 1: u = x @ Uall + bUall via bf16 WMMA ----------------
  v8f acc[7][2];
  #pragma unroll
  for (int mt = 0; mt < 7; ++mt) {
    #pragma unroll
    for (int nt = 0; nt < 2; ++nt) {
      v8f z = {0.f,0.f,0.f,0.f,0.f,0.f,0.f,0.f};
      acc[mt][nt] = z;
    }
  }

  for (int kc = 0; kc < E_; kc += KC) {
    __syncthreads();
    // ---- stage A slice into fragment-major layout (pairs along K) ----
    // pairs: [112 rows][64 k-pairs]; rows >= 100 zero-padded
    for (int i = tid; i < MPAD*(KC/2); i += NTHREADS) {
      int r  = i >> 6, cp = i & 63;
      int k  = 2*cp;
      float v0 = 0.f, v1 = 0.f;
      if (r < L_*T_) {
        v0 = xS[r*E_ + kc + k];
        v1 = xS[r*E_ + kc + k + 1];
      }
      int mt   = r >> 4, m_in = r & 15;
      int kk   = k >> 5, k_in = k & 31;
      int hl   = (k_in >> 3) & 1;
      int e    = ((k_in >> 4) << 3) + (k_in & 7);      // even
      int lane_a = hl*16 + m_in;
      As32[(((kk*7 + mt)*32 + lane_a) << 3) + (e >> 1)] = pack_bf16x2(v0, v1);
    }
    // ---- stage B slice into fragment-major layout (pairs along K) ----
    // pairs: [64 k-pairs][256 n]; n fastest for coalesced global reads
    for (int i = tid; i < (KC/2)*G_; i += NTHREADS) {
      int n  = i & (G_-1), kp = i >> 8;
      int k  = 2*kp;
      float v0 = uS[(size_t)(kc + k  )*G_ + n];
      float v1 = uS[(size_t)(kc + k+1)*G_ + n];
      int kk = k >> 5, k_in = k & 31;
      int hl = (k_in >> 3) & 1;
      int e  = ((k_in >> 4) << 3) + (k_in & 7);        // even
      int nt = n >> 4, nl = n & 15;
      int lane_b = hl*16 + nl;
      Bs32[(((kk*16 + nt)*32 + lane_b) << 3) + (e >> 1)] = pack_bf16x2(v0, v1);
    }
    // ---- prefetch next chunk while this one computes ----
    if (kc + KC < E_) {
      for (int i = tid; i < 1024; i += NTHREADS)   // Uall next chunk: 128 rows x 8 lines
        __builtin_prefetch(&uS[(size_t)(kc + KC + (i >> 3))*G_ + (i & 7)*32], 0, 0);
      for (int i = tid; i < 400; i += NTHREADS)    // x next chunk: 100 rows x 4 lines
        __builtin_prefetch(&xS[(i >> 2)*E_ + kc + KC + (i & 3)*32], 0, 0);
    }
    __syncthreads();

    #pragma unroll
    for (int kk = 0; kk < KC/32; ++kk) {
      v16bf fb0 = Bfrag[(kk*16 + wave*2    )*32 + lane];
      v16bf fb1 = Bfrag[(kk*16 + wave*2 + 1)*32 + lane];
      #pragma unroll
      for (int mt = 0; mt < 7; ++mt) {
        v16bf fa = Afrag[(kk*7 + mt)*32 + lane];
        acc[mt][0] = __builtin_amdgcn_wmma_f32_16x16x32_bf16(
            false, fa, false, fb0, (short)0, acc[mt][0], false, false);
        acc[mt][1] = __builtin_amdgcn_wmma_f32_16x16x32_bf16(
            false, fa, false, fb1, (short)0, acc[mt][1], false, false);
      }
    }
  }
  // store D tiles into u (fold bUall in). C/D layout: VGPR r -> row r (+8 for hi lanes)
  {
    const int nl   = lane & 15;
    const int rofs = (lane >> 4) * 8;
    #pragma unroll
    for (int mt = 0; mt < 7; ++mt) {
      #pragma unroll
      for (int nt = 0; nt < 2; ++nt) {
        int col = wave*32 + nt*16 + nl;
        #pragma unroll
        for (int r = 0; r < 8; ++r) {
          int row = mt*16 + rofs + r;
          u_s[row*G_ + col] = acc[mt][nt][r] + bU_s[col];
        }
      }
    }
  }
  __syncthreads();

  // ---------------- Phase 2: TimeLSTM scan over T ----------------
  float* Wall_s  = (float*)(smem + OFF_WALL);
  float* Wd_s    = (float*)(smem + OFF_WD);
  float* bWall_s = (float*)(smem + OFF_BWALL);
  float* bWd_s   = (float*)(smem + OFF_BWD);
  float* ts_s    = (float*)(smem + OFF_TS);
  float* h_s     = (float*)(smem + OFF_H);
  float* c_s     = (float*)(smem + OFF_C);
  float* cs1_s   = (float*)(smem + OFF_CS1);
  float* gates_s = (float*)(smem + OFF_GATES);
  float* full_s  = (float*)(smem + OFF_FULL);

  for (int i = tid; i < H_*G_; i += NTHREADS) Wall_s[i] = p_Wall[(size_t)s*H_*G_ + i];
  for (int i = tid; i < H_*H_; i += NTHREADS) Wd_s[i]   = p_Wd[(size_t)s*H_*H_ + i];
  for (int i = tid; i < G_;    i += NTHREADS) bWall_s[i]= p_bWall[s*G_ + i];
  for (int i = tid; i < H_;    i += NTHREADS) bWd_s[i]  = p_bWd[s*H_ + i];
  for (int i = tid; i < L_*T_; i += NTHREADS) ts_s[i]   = p_ts[s*L_*T_ + i];
  for (int i = tid; i < L_*H_; i += NTHREADS) { h_s[i] = 0.f; c_s[i] = 0.f; }
  __syncthreads();

  for (int t = 0; t < T_; ++t) {
    // c_s1 = tanh(c @ Wd + bWd)
    for (int i = tid; i < L_*H_; i += NTHREADS) {
      int l = i >> 6, j = i & 63;
      float a = bWd_s[j];
      for (int k = 0; k < H_; ++k) a += c_s[l*H_ + k] * Wd_s[k*H_ + j];
      cs1_s[i] = tanhf(a);
    }
    // gates = h @ Wall + bWall + u_t
    for (int i = tid; i < L_*G_; i += NTHREADS) {
      int l = i >> 8, g = i & 255;
      float a = bWall_s[g] + u_s[(l*T_ + t)*G_ + g];
      for (int k = 0; k < H_; ++k) a += h_s[l*H_ + k] * Wall_s[k*G_ + g];
      gates_s[i] = a;
    }
    __syncthreads();
    for (int i = tid; i < L_*H_; i += NTHREADS) {
      int l = i >> 6, j = i & 63;
      float gf = sigm(gates_s[l*G_ + j]);
      float gi = sigm(gates_s[l*G_ + 64 + j]);
      float go = sigm(gates_s[l*G_ + 128 + j]);
      float gc = sigm(gates_s[l*G_ + 192 + j]);  // sigmoid, faithful to source
      float cs = cs1_s[i];
      float cadj = (c_s[i] - cs) + cs * ts_s[l*T_ + t];
      float cn = gf*cadj + gi*gc;
      c_s[i] = cn;
      h_s[i] = go * tanhf(cn);
      full_s[(t*L_ + l)*H_ + j] = go;
    }
    __syncthreads();
  }

  // ---------------- Phase 3: text attention ----------------
  float* W1_s   = (float*)(smem + OFF_W1);
  float* W2_s   = (float*)(smem + OFF_W2);
  float* b1_s   = (float*)(smem + OFF_B1);
  float* b2_s   = (float*)(smem + OFF_B2);
  float* V_s    = (float*)(smem + OFF_V);
  float* w1h_s  = (float*)(smem + OFF_W1H);
  float* sc_s   = (float*)(smem + OFF_SCORE);
  float* tvec_s = (float*)(smem + OFF_TVEC);

  for (int i = tid; i < H_*H_; i += NTHREADS) {
    W1_s[i] = p_W1[(size_t)s*H_*H_ + i];
    W2_s[i] = p_W2[(size_t)s*H_*H_ + i];
  }
  for (int i = tid; i < H_; i += NTHREADS) {
    b1_s[i] = p_b1[s*H_ + i]; b2_s[i] = p_b2[s*H_ + i]; V_s[i] = p_V[s*H_ + i];
  }
  __syncthreads();
  const float bV_v = p_bV[s];

  // W1h[l] = h_fin[l] @ W1 + b1 + b2
  for (int i = tid; i < L_*H_; i += NTHREADS) {
    int l = i >> 6, j = i & 63;
    float a = b1_s[j] + b2_s[j];
    for (int k = 0; k < H_; ++k) a += h_s[l*H_ + k] * W1_s[k*H_ + j];
    w1h_s[i] = a;
  }
  __syncthreads();
  if (tid < L_*T_) {
    int l = tid / T_, t = tid % T_;
    float a = 0.f;
    for (int j = 0; j < H_; ++j) {
      float tmp = w1h_s[l*H_ + j];
      for (int k = 0; k < H_; ++k) tmp += full_s[(t*L_ + l)*H_ + k] * W2_s[k*H_ + j];
      a += tanhf(tmp) * V_s[j];
    }
    sc_s[l*T_ + t] = a + bV_v;
  }
  __syncthreads();
  if (tid < L_) {
    int l = tid;
    float mx = -1e30f;
    for (int t = 0; t < T_; ++t) mx = fmaxf(mx, sc_s[l*T_ + t]);
    float sum = 0.f;
    for (int t = 0; t < T_; ++t) { float e = __expf(sc_s[l*T_ + t] - mx); sc_s[l*T_ + t] = e; sum += e; }
    float inv = 1.f / sum;
    for (int t = 0; t < T_; ++t) sc_s[l*T_ + t] *= inv;
  }
  __syncthreads();
  for (int i = tid; i < L_*H_; i += NTHREADS) {
    int l = i >> 6, j = i & 63;
    float a = 0.f;
    for (int t = 0; t < T_; ++t) a += sc_s[l*T_ + t] * full_s[(t*L_ + l)*H_ + j];
    tvec_s[i] = a;
  }
  __syncthreads();

  // ---------------- Phase 4: day LSTM (one step, zero init; h0@Whh == 0) ----------------
  float* Wih_s  = (float*)(smem + OFF_WIH);
  float* bsum_s = (float*)(smem + OFF_BSUM);
  float* hd_s   = (float*)(smem + OFF_HD);

  for (int i = tid; i < H_*G_; i += NTHREADS) Wih_s[i] = p_Wih[(size_t)s*H_*G_ + i];
  for (int i = tid; i < G_;    i += NTHREADS) bsum_s[i] = p_bih[s*G_ + i] + p_bhh[s*G_ + i];
  __syncthreads();
  for (int i = tid; i < L_*G_; i += NTHREADS) {
    int l = i >> 8, g = i & 255;
    float a = bsum_s[g];
    for (int k = 0; k < H_; ++k) a += tvec_s[l*H_ + k] * Wih_s[k*G_ + g];
    gates_s[i] = a;
  }
  __syncthreads();
  for (int i = tid; i < L_*H_; i += NTHREADS) {
    int l = i >> 6, j = i & 63;
    float gi = sigm(gates_s[l*G_ + j]);
    float gg = tanhf(gates_s[l*G_ + 128 + j]);
    float go = sigm(gates_s[l*G_ + 192 + j]);
    float cd = gi * gg;
    hd_s[i] = go * tanhf(cd);
  }
  __syncthreads();

  // ---------------- Phase 5: day attention + head ----------------
  float* s2_s = (float*)(smem + OFF_S2);   // s2[0..4], aw2 at [8..12]
  float* ft_s = (float*)(smem + OFF_FT);

  if (tid < L_) {
    int l = tid;
    float a = 0.f;
    for (int j = 0; j < H_; ++j) {
      float tmp = b1_s[j] + b2_s[j];
      for (int k = 0; k < H_; ++k) tmp += hd_s[l*H_ + k] * (W1_s[k*H_ + j] + W2_s[k*H_ + j]);
      a += tanhf(tmp) * V_s[j];
    }
    s2_s[l] = a + bV_v;
  }
  __syncthreads();
  if (tid == 0) {
    float mx = -1e30f;
    for (int l = 0; l < L_; ++l) mx = fmaxf(mx, s2_s[l]);
    float sum = 0.f;
    for (int l = 0; l < L_; ++l) { float e = __expf(s2_s[l] - mx); s2_s[8 + l] = e; sum += e; }
    float inv = 1.f / sum;
    for (int l = 0; l < L_; ++l) s2_s[8 + l] *= inv;
  }
  __syncthreads();
  if (tid < H_) {
    float a = 0.f;
    for (int l = 0; l < L_; ++l) a += s2_s[8 + l] * hd_s[l*H_ + tid];
    ft_s[tid] = a;
  }
  __syncthreads();
  if (tid == 0) {
    float o = p_linb[0];
    for (int j = 0; j < H_; ++j) o += ft_s[j] * p_linw[j];
    p_out[s] = (o > 0.f) ? o : 0.01f * o;
  }
}

extern "C" void kernel_launch(void* const* d_in, const int* in_sizes, int n_in,
                              void* d_out, int out_size, void* d_ws, size_t ws_size,
                              hipStream_t stream) {
  (void)in_sizes; (void)n_in; (void)d_ws; (void)ws_size; (void)out_size;
  const float* x     = (const float*)d_in[0];
  const float* ts    = (const float*)d_in[1];
  const float* Wall  = (const float*)d_in[2];
  const float* bWall = (const float*)d_in[3];
  const float* Uall  = (const float*)d_in[4];
  const float* bUall = (const float*)d_in[5];
  const float* Wd    = (const float*)d_in[6];
  const float* bWd   = (const float*)d_in[7];
  const float* W1    = (const float*)d_in[8];
  const float* b1    = (const float*)d_in[9];
  const float* W2    = (const float*)d_in[10];
  const float* b2    = (const float*)d_in[11];
  const float* V     = (const float*)d_in[12];
  const float* bV    = (const float*)d_in[13];
  const float* Wih   = (const float*)d_in[14];
  // d_in[15] = Whh, unused (h0 == 0)
  const float* bih   = (const float*)d_in[16];
  const float* bhh   = (const float*)d_in[17];
  const float* linw  = (const float*)d_in[18];
  const float* linb  = (const float*)d_in[19];
  float* out = (float*)d_out;

  hipLaunchKernelGGL(athena_stocknet_kernel, dim3(S_), dim3(NTHREADS), SMEM_BYTES, stream,
                     x, ts, Wall, bWall, Uall, bUall, Wd, bWd,
                     W1, b1, W2, b2, V, bV, Wih, bih, bhh, linw, linb, out);
}
